// GATQuestionGuidedCross_38482906972433
// MI455X (gfx1250) — matrix operand order repaired and checked by reading, not compiled, not run on hardware
//
#include <hip/hip_runtime.h>
#include <stdint.h>

// Problem constants (match reference)
#define BB    128   // batch
#define NIMG  128
#define NKG   256
#define DQ    1024  // question dim
#define DM    512   // model dim D
#define HN    8     // heads
#define DHD   64    // head dim
#define MIDD  2048  // FFN mid

typedef __attribute__((ext_vector_type(16))) __bf16        v16bf;
typedef __attribute__((ext_vector_type(8)))  float         v8f;
typedef __attribute__((ext_vector_type(4)))  int           v4i;
typedef __attribute__((ext_vector_type(4)))  unsigned int  v4u;
typedef __attribute__((ext_vector_type(8)))  int           v8i_;

union BF16x16 { v16bf v; unsigned short u[16]; uint4 q[2]; };
union F32x8   { v8f v; float f[8]; };

__device__ __forceinline__ unsigned short f2bf(float f) {
  unsigned int u = __float_as_uint(f);
  u += 0x7FFFu + ((u >> 16) & 1u);   // round-to-nearest-even
  return (unsigned short)(u >> 16);
}

// ---------------------------------------------------------------------------
// CDNA5 async global->LDS copy (ASYNCcnt-tracked).
// Builtin signature: (v4i addrspace(1)*, v4i addrspace(3)*, imm off, imm cpol)
// ---------------------------------------------------------------------------
#if defined(__has_builtin)
#if __has_builtin(__builtin_amdgcn_global_load_async_to_lds_b128)
#define HAVE_ASYNC_LDS 1
#endif
#if __has_builtin(__builtin_amdgcn_tensor_load_to_lds)
#define HAVE_TDM 1
#endif
#endif

#if HAVE_ASYNC_LDS
__device__ __forceinline__ void async_copy16(void* lds, const void* glob) {
  __builtin_amdgcn_global_load_async_to_lds_b128(
      (__attribute__((address_space(1))) v4i*)glob,
      (__attribute__((address_space(3))) v4i*)lds,
      0, 0);
}
__device__ __forceinline__ void async_wait0() {
#if __has_builtin(__builtin_amdgcn_s_wait_asynccnt)
  __builtin_amdgcn_s_wait_asynccnt(0);
#else
  asm volatile("s_wait_asynccnt 0" ::: "memory");
#endif
}
#else
__device__ __forceinline__ void async_copy16(void* lds, const void* glob) {
  *(uint4*)lds = *(const uint4*)glob;
}
__device__ __forceinline__ void async_wait0() {}
#endif

// ---------------------------------------------------------------------------
// TDM: one 2-D tensor_load_to_lds covering a [rows=128, k=32] bf16 tile with
// LDS row padding (pad_interval = 16 DWORDs, pad_amount = 4 DWORDs -> 80B
// rows) for bank-conflict-free WMMA fragment reads. D# packed per ISA §8.
// This toolchain's builtin is the 6-arg form:
//   (uint32x4 g0, int32x8 g1, int32x4 g2, int32x4 g3, int32x8 extra, i32 cpol)
// ---------------------------------------------------------------------------
#if HAVE_TDM
typedef __attribute__((address_space(3))) const void* as3_cvp;
__device__ __forceinline__ unsigned lds_addr_of(const void* p) {
  return (unsigned)(unsigned long long)(as3_cvp)p;
}
__device__ __forceinline__ void tdm_load_tile(
    unsigned ldsAddr, const void* gsrc, int Kelems, int Nrows)
{
  unsigned long long ga = (unsigned long long)(size_t)gsrc;
  v4u g0;
  g0.x = 1u;                                             // count=1, user mode
  g0.y = ldsAddr;                                        // lds_addr (bytes)
  g0.z = (unsigned)(ga & 0xFFFFFFFFu);                   // global_addr lo
  g0.w = (unsigned)((ga >> 32) & 0x01FFFFFFu) | (2u << 30); // addr hi | type=2
  v8i_ g1;
  g1[0] = (1 << 16)      // data_size = 1 -> 2 bytes
        | (1 << 20)      // pad_enable
        | (3 << 22)      // pad_interval: 2^(3+1) = 16 DWORDs (= 64B row)
        | (3 << 25);     // pad_amount: 3+1 = 4 DWORDs (= 16B pad -> 80B rows)
  g1[1] = (Kelems & 0xFFFF) << 16;                       // tensor_dim0 lo16
  g1[2] = ((Kelems >> 16) & 0xFFFF) | ((Nrows & 0xFFFF) << 16); // dim0 hi | dim1 lo
  g1[3] = ((Nrows >> 16) & 0xFFFF) | (32 << 16);         // dim1 hi | tile_dim0=32
  g1[4] = 128;                                           // tile_dim1=128, tile_dim2=0
  g1[5] = Kelems;                                        // tensor_dim0_stride lo32
  g1[6] = 0;                                             // stride hi | dim1_stride lo
  g1[7] = 0;
  v4i z4 = {};
  v8i_ z8 = {};
  __builtin_amdgcn_tensor_load_to_lds(g0, g1, z4, z4, z8, 0);
}
__device__ __forceinline__ void tensor_wait0() {
#if __has_builtin(__builtin_amdgcn_s_wait_tensorcnt)
  __builtin_amdgcn_s_wait_tensorcnt(0);
#else
  asm volatile("s_wait_tensorcnt 0x0" ::: "memory");
#endif
}
#endif

// ---------------------------------------------------------------------------
// fp32 -> bf16 conversion (activations), grid-stride
// ---------------------------------------------------------------------------
__global__ void k_cvt_bf16(const float* __restrict__ s, unsigned short* __restrict__ d, int n) {
  int i = blockIdx.x * blockDim.x + threadIdx.x;
  int stride = gridDim.x * blockDim.x;
  for (; i < n; i += stride) d[i] = f2bf(s[i]);
}

// ---------------------------------------------------------------------------
// fp32 [K,N] -> bf16 [N,K] transpose-convert (one-time weight preprocessing).
// ---------------------------------------------------------------------------
__global__ __launch_bounds__(256) void k_cvtT_bf16(
    const float* __restrict__ s, unsigned short* __restrict__ d, int Kd, int Nd)
{
  __shared__ float t[32][33];
  const int kb = blockIdx.y * 32, nb = blockIdx.x * 32;
  const int tx = threadIdx.x & 31, ty = threadIdx.x >> 5;   // 32 x 8
#pragma unroll
  for (int i = 0; i < 32; i += 8)
    t[ty + i][tx] = s[(size_t)(kb + ty + i) * Nd + (nb + tx)];
  __syncthreads();
#pragma unroll
  for (int i = 0; i < 32; i += 8)
    d[(size_t)(nb + ty + i) * Kd + (kb + tx)] = f2bf(t[tx][ty + i]);
}

// ---------------------------------------------------------------------------
// Tiled WMMA GEMM: C[M,N] = act(A[M,K](bf16) @ Bt[N,K](bf16)^T + bias)
// Block tile 128x128, K-step 32, 8 waves, 2x4 WMMA tiles per wave (32x64).
// A tile: per-thread async global->LDS copies. B tile: single TDM
// tensor_load_to_lds (wave 0) with pad-to-80B rows; async fallback.
// LDS rows padded to 40 ushorts: fragment reads stride 20 banks -> conflict-free.
// M,N multiples of 128; K multiple of 32.
// ---------------------------------------------------------------------------
__global__ __launch_bounds__(256) void k_gemm_bf16(
    const unsigned short* __restrict__ A,
    const unsigned short* __restrict__ Bt,    // [N,K] pre-transposed
    const float* __restrict__ bias,
    float* __restrict__ Cf,
    unsigned short* __restrict__ Cb,
    int M, int N, int K, int relu)
{
  __shared__ __align__(16) unsigned short As[2][128][40];   // [buf][m][k] (+pad)
  __shared__ __align__(16) unsigned short Bs[2][128][40];   // [buf][n][k] (+pad)
  const int tid  = threadIdx.x;
  const int lane = tid & 31;
  const int wv   = tid >> 5;
  const int tileM = blockIdx.y * 128;
  const int tileN = blockIdx.x * 128;
  const int wm = (wv >> 1) * 32;     // 4 row groups of 32
  const int wn = (wv & 1) * 64;      // 2 col groups of 64
  const int hf  = lane >> 4;
  const int l15 = lane & 15;

  // staging assignment: thread -> (row r, k-chunk c), 32B per thread per tile
  const int r = tid >> 1;            // 0..127
  const int c = (tid & 1) << 4;      // 0 or 16
  const unsigned short* aRow = A  + (size_t)(tileM + r) * K + c;
#if !HAVE_TDM
  const unsigned short* bRow = Bt + (size_t)(tileN + r) * K + c;
#endif

  v8f acc[2][4] = {};

  // prologue: stage tile 0 into buffer 0
  async_copy16(&As[0][r][c],     aRow);
  async_copy16(&As[0][r][c + 8], aRow + 8);
#if HAVE_TDM
  const unsigned ldsB0 = lds_addr_of(&Bs[0][0][0]);
  if (wv == 0)
    tdm_load_tile(ldsB0, Bt + (size_t)tileN * K, K, N);
#else
  async_copy16(&Bs[0][r][c],     bRow);
  async_copy16(&Bs[0][r][c + 8], bRow + 8);
#endif

  for (int k0 = 0; k0 < K; k0 += 32) {
    const int cur = (k0 >> 5) & 1;
    async_wait0();
#if HAVE_TDM
    if (wv == 0) tensor_wait0();
#endif
    __syncthreads();                 // everyone's copies for `cur` are visible
    if (k0 + 32 < K) {               // stage next tile into the other buffer
      const int nxt = cur ^ 1;
      async_copy16(&As[nxt][r][c],     aRow + k0 + 32);
      async_copy16(&As[nxt][r][c + 8], aRow + k0 + 40);
#if HAVE_TDM
      if (wv == 0)
        tdm_load_tile(ldsB0 + (unsigned)(nxt * (int)sizeof(Bs[0])),
                      Bt + (size_t)tileN * K + k0 + 32, K, N);
#else
      async_copy16(&Bs[nxt][r][c],     bRow + k0 + 32);
      async_copy16(&Bs[nxt][r][c + 8], bRow + k0 + 40);
#endif
    }

    // ---- fragments (CDNA5 16-bit WMMA layouts) ----
    BF16x16 af[2], bf16v[4];
#pragma unroll
    for (int mi = 0; mi < 2; ++mi) {
      // A 16x32: lane holds row=l15, K = hf*16 .. +15 contiguous
      int rr = wm + mi * 16 + l15;
      int kb = hf << 4;
      af[mi].q[0] = *(const uint4*)&As[cur][rr][kb];
      af[mi].q[1] = *(const uint4*)&As[cur][rr][kb + 8];
    }
#pragma unroll
    for (int ni = 0; ni < 4; ++ni) {
      // B 32x16: lane holds col=l15, K = hf*16 .. +15 contiguous
      int cc = wn + ni * 16 + l15;
      int ks = hf << 4;
      bf16v[ni].q[0] = *(const uint4*)&Bs[cur][cc][ks];
      bf16v[ni].q[1] = *(const uint4*)&Bs[cur][cc][ks + 8];
    }
#pragma unroll
    for (int mi = 0; mi < 2; ++mi)
#pragma unroll
      for (int ni = 0; ni < 4; ++ni)
        acc[mi][ni] = __builtin_amdgcn_wmma_f32_16x16x32_bf16(
            false, af[mi].v, false, bf16v[ni].v, (short)0, acc[mi][ni], false, false);
  }

  // ---- epilogue: C layout VGPR j -> row = j + 8*hf, col = l15 ----
#pragma unroll
  for (int mi = 0; mi < 2; ++mi)
#pragma unroll
    for (int ni = 0; ni < 4; ++ni) {
      F32x8 rr; rr.v = acc[mi][ni];
      int col = tileN + wn + ni * 16 + l15;
      float bv = bias ? bias[col] : 0.f;
#pragma unroll
      for (int j = 0; j < 8; ++j) {
        int row = tileM + wm + mi * 16 + (hf << 3) + j;
        float vv = rr.f[j] + bv;
        if (relu) vv = fmaxf(vv, 0.f);
        size_t idx = (size_t)row * N + col;
        if (Cf) Cf[idx] = vv;
        if (Cb) Cb[idx] = f2bf(vv);
      }
    }
}

// ---------------------------------------------------------------------------
// Flash attention: one wave per (b, h, 16-query tile). Keys chunked by 32.
// Q,K,V,O stored [B, Nrows, H*DH] bf16. Scale = 1/sqrt(64).
// ---------------------------------------------------------------------------
__global__ __launch_bounds__(32) void k_attn(
    const unsigned short* __restrict__ Qb,
    const unsigned short* __restrict__ Kb,
    const unsigned short* __restrict__ Vb,
    unsigned short* __restrict__ Ob,
    int Nq, int Nk)
{
  __shared__ __align__(16) unsigned short Ps[16][32];
  const int lane = threadIdx.x;
  const int hf = lane >> 4, l15 = lane & 15;
  const int qt = blockIdx.x * 16;
  const int h  = blockIdx.y;
  const int b  = blockIdx.z;
  const size_t rs = (size_t)HN * DHD;   // 512
  const unsigned short* Qp = Qb + ((size_t)b * Nq) * rs + (size_t)h * DHD;
  const unsigned short* Kp = Kb + ((size_t)b * Nk) * rs + (size_t)h * DHD;
  const unsigned short* Vp = Vb + ((size_t)b * Nk) * rs + (size_t)h * DHD;

  BF16x16 qa[2];
#pragma unroll
  for (int s = 0; s < 2; ++s) {
    const unsigned short* src = Qp + (size_t)(qt + l15) * rs + s * 32 + hf * 16;
    qa[s].q[0] = *(const uint4*)src;
    qa[s].q[1] = *(const uint4*)(src + 8);
  }

  float mrun[8], lrun[8];
  v8f o[4] = {};
#pragma unroll
  for (int j = 0; j < 8; ++j) { mrun[j] = -3.0e38f; lrun[j] = 0.f; }
  const float scale = 0.125f;

  for (int kt = 0; kt < Nk; kt += 32) {
    float svv[2][8];
#pragma unroll
    for (int sub = 0; sub < 2; ++sub) {
      const int keyBase = kt + sub * 16;
      v8f sa = {};
#pragma unroll
      for (int s = 0; s < 2; ++s) {
        BF16x16 kb16;  // B operand: lane=key col, K=d contiguous per half
        const unsigned short* src = Kp + (size_t)(keyBase + l15) * rs + s * 32 + hf * 16;
        kb16.q[0] = *(const uint4*)src;
        kb16.q[1] = *(const uint4*)(src + 8);
        sa = __builtin_amdgcn_wmma_f32_16x16x32_bf16(
            false, qa[s].v, false, kb16.v, (short)0, sa, false, false);
      }
      F32x8 sr; sr.v = sa;
#pragma unroll
      for (int j = 0; j < 8; ++j) svv[sub][j] = sr.f[j] * scale;
    }
    float mnew[8], alpha[8], rsum[8];
#pragma unroll
    for (int j = 0; j < 8; ++j) {
      float mx = fmaxf(svv[0][j], svv[1][j]);
#pragma unroll
      for (int d = 1; d < 16; d <<= 1) mx = fmaxf(mx, __shfl_xor(mx, d, 32));
      mnew[j]  = fmaxf(mrun[j], mx);
      alpha[j] = __expf(mrun[j] - mnew[j]);
      rsum[j]  = 0.f;
    }
#pragma unroll
    for (int sub = 0; sub < 2; ++sub)
#pragma unroll
      for (int j = 0; j < 8; ++j) {
        float p = __expf(svv[sub][j] - mnew[j]);
        rsum[j] += p;
        Ps[j + hf * 8][sub * 16 + l15] = f2bf(p);   // C-layout -> LDS row-major
      }
#pragma unroll
    for (int j = 0; j < 8; ++j) {
      float s = rsum[j];
#pragma unroll
      for (int d = 1; d < 16; d <<= 1) s += __shfl_xor(s, d, 32);
      lrun[j] = lrun[j] * alpha[j] + s;
      mrun[j] = mnew[j];
    }
    __syncthreads();
    BF16x16 pa;   // reload P as A-operand (16x32)
    pa.q[0] = *(const uint4*)&Ps[l15][hf * 16];
    pa.q[1] = *(const uint4*)&Ps[l15][hf * 16 + 8];
#pragma unroll
    for (int t = 0; t < 4; ++t) {
      F32x8 ov; ov.v = o[t];
#pragma unroll
      for (int j = 0; j < 8; ++j) ov.f[j] *= alpha[j];
      BF16x16 vb16;  // B operand: lane=d col, K=key rows
#pragma unroll
      for (int i = 0; i < 16; ++i)
        vb16.u[i] = Vp[(size_t)(kt + hf * 16 + i) * rs + t * 16 + l15];
      o[t] = __builtin_amdgcn_wmma_f32_16x16x32_bf16(
          false, pa.v, false, vb16.v, (short)0, ov.v, false, false);
    }
    __syncthreads();
  }
#pragma unroll
  for (int t = 0; t < 4; ++t) {
    F32x8 ov; ov.v = o[t];
#pragma unroll
    for (int j = 0; j < 8; ++j) {
      const int row = qt + j + (hf << 3);
      const int col = t * 16 + l15;
      Ob[((size_t)b * Nq + row) * rs + (size_t)h * DHD + col] = f2bf(ov.f[j] / lrun[j]);
    }
  }
}

// ---------------------------------------------------------------------------
// LayerNorm over D=512 of (Xa + Xb); one wave per row, 8 rows per block.
// ---------------------------------------------------------------------------
__global__ __launch_bounds__(256) void k_ln512(
    const float* __restrict__ Xa, const float* __restrict__ Xb,
    const float* __restrict__ g, const float* __restrict__ be,
    float* __restrict__ outF, unsigned short* __restrict__ outB)
{
  const int lane = threadIdx.x & 31;
  const int wv = threadIdx.x >> 5;
  const size_t row = (size_t)blockIdx.x * 8 + wv;
  const float* pa = Xa + row * DM;
  const float* pb = Xb + row * DM;
  float v[16]; float s = 0.f;
#pragma unroll
  for (int i = 0; i < 16; ++i) { int c = lane + i * 32; v[i] = pa[c] + pb[c]; s += v[i]; }
#pragma unroll
  for (int d = 1; d < 32; d <<= 1) s += __shfl_xor(s, d, 32);
  float mean = s * (1.f / 512.f);
  float var = 0.f;
#pragma unroll
  for (int i = 0; i < 16; ++i) { float dd = v[i] - mean; var += dd * dd; }
#pragma unroll
  for (int d = 1; d < 32; d <<= 1) var += __shfl_xor(var, d, 32);
  float inv = rsqrtf(var * (1.f / 512.f) + 1e-5f);
#pragma unroll
  for (int i = 0; i < 16; ++i) {
    int c = lane + i * 32;
    float y = (v[i] - mean) * inv * g[c] + be[c];
    outF[row * DM + c] = y;
    if (outB) outB[row * DM + c] = f2bf(y);
  }
}

// ---------------------------------------------------------------------------
// feat = tanh(qvec[b,:] + nodes[row,:]) -> f32 + bf16
// ---------------------------------------------------------------------------
__global__ __launch_bounds__(256) void k_feat_tanh(
    const float* __restrict__ nodes, const float* __restrict__ qv,
    float* __restrict__ outF, unsigned short* __restrict__ outB, int Nn)
{
  const int lane = threadIdx.x & 31;
  const int wv = threadIdx.x >> 5;
  const size_t row = (size_t)blockIdx.x * 8 + wv;
  const int b = (int)(row / (size_t)Nn);
#pragma unroll
  for (int i = 0; i < 16; ++i) {
    int c = lane + i * 32;
    float x = tanhf(qv[(size_t)b * DM + c] + nodes[row * DM + c]);
    outF[row * DM + c] = x;
    outB[row * DM + c] = f2bf(x);
  }
}

// ---------------------------------------------------------------------------
// final: dst[b, rowOff+n, :] = l2norm(nodes[row,:] + blk[row,:])
// ---------------------------------------------------------------------------
__global__ __launch_bounds__(256) void k_res_l2(
    const float* __restrict__ nodes, const float* __restrict__ blk,
    float* __restrict__ dst, int Nn, int rowOff)
{
  const int lane = threadIdx.x & 31;
  const int wv = threadIdx.x >> 5;
  const size_t row = (size_t)blockIdx.x * 8 + wv;
  const int b = (int)(row / (size_t)Nn);
  const int n = (int)(row - (size_t)b * Nn);
  float v[16]; float ss = 0.f;
#pragma unroll
  for (int i = 0; i < 16; ++i) {
    int c = lane + i * 32;
    v[i] = nodes[row * DM + c] + blk[row * DM + c];
    ss += v[i] * v[i];
  }
#pragma unroll
  for (int d = 1; d < 32; d <<= 1) ss += __shfl_xor(ss, d, 32);
  float sc = 1.f / fmaxf(sqrtf(ss), 1e-12f);
  size_t drow = (size_t)b * (NIMG + NKG) + rowOff + n;
#pragma unroll
  for (int i = 0; i < 16; ++i) dst[drow * DM + lane + i * 32] = v[i] * sc;
}

// ---------------------------------------------------------------------------
// Host
// ---------------------------------------------------------------------------
extern "C" void kernel_launch(void* const* d_in, const int* in_sizes, int n_in,
                              void* d_out, int out_size, void* d_ws, size_t ws_size,
                              hipStream_t stream)
{
  (void)in_sizes; (void)n_in; (void)out_size; (void)ws_size;

  // Input flattening: top-level in setup_inputs() order; nested params in
  // jax pytree (sorted-key) order. qg_* params are mathematically unused
  // (softmax over a singleton axis == 1.0 -> qg_att is identity).
  const float* question  = (const float*)d_in[0];
  const float* img_nodes = (const float*)d_in[1];
  const float* kg_nodes  = (const float*)d_in[2];
  const float* Wqi = (const float*)d_in[3];
  const float* Wqk = (const float*)d_in[4];
  const float* bqi = (const float*)d_in[5];
  const float* bqk = (const float*)d_in[6];

  struct CrossP {
    const float *W1,*W2,*Wk,*Wo,*Wq,*Wv,*b1,*b2,*bk,*bo,*bq,*bv,*ln1b,*ln1g,*ln2b,*ln2g;
  };
  auto loadCross = [&](int base) {
    CrossP c;
    c.W1 = (const float*)d_in[base + 0];  c.W2 = (const float*)d_in[base + 1];
    c.Wk = (const float*)d_in[base + 2];  c.Wo = (const float*)d_in[base + 3];
    c.Wq = (const float*)d_in[base + 4];  c.Wv = (const float*)d_in[base + 5];
    c.b1 = (const float*)d_in[base + 6];  c.b2 = (const float*)d_in[base + 7];
    c.bk = (const float*)d_in[base + 8];  c.bo = (const float*)d_in[base + 9];
    c.bq = (const float*)d_in[base + 10]; c.bv = (const float*)d_in[base + 11];
    c.ln1b = (const float*)d_in[base + 12]; c.ln1g = (const float*)d_in[base + 13];
    c.ln2b = (const float*)d_in[base + 14]; c.ln2g = (const float*)d_in[base + 15];
    return c;
  };
  CrossP i2k = loadCross(7), k2i = loadCross(23);

  // ---- workspace layout ----
  char* ws = (char*)d_ws;
  size_t off = 0;
  auto alloc = [&](size_t bytes) -> void* {
    void* p = ws + off;
    off += (bytes + 255) & ~(size_t)255;
    return p;
  };
  const int Mimg = BB * NIMG;   // 16384
  const int Mkg  = BB * NKG;    // 32768

  unsigned short* q16   = (unsigned short*)alloc((size_t)BB * DQ * 2);
  unsigned short* wqi16 = (unsigned short*)alloc((size_t)DQ * DM * 2);   // [N,K]
  unsigned short* wqk16 = (unsigned short*)alloc((size_t)DQ * DM * 2);
  struct CW { unsigned short *q,*k,*v,*o,*w1,*w2; };   // all [N,K] transposed bf16
  auto allocCW = [&]() {
    CW w;
    w.q  = (unsigned short*)alloc((size_t)DM * DM * 2);
    w.k  = (unsigned short*)alloc((size_t)DM * DM * 2);
    w.v  = (unsigned short*)alloc((size_t)DM * DM * 2);
    w.o  = (unsigned short*)alloc((size_t)DM * DM * 2);
    w.w1 = (unsigned short*)alloc((size_t)DM * MIDD * 2);
    w.w2 = (unsigned short*)alloc((size_t)MIDD * DM * 2);
    return w;
  };
  CW wi2k = allocCW(), wk2i = allocCW();

  float* qimgF = (float*)alloc((size_t)BB * DM * 4);
  float* qkgF  = (float*)alloc((size_t)BB * DM * 4);
  float*          imgFeatF = (float*)alloc((size_t)Mimg * DM * 4);
  unsigned short* imgFeatB = (unsigned short*)alloc((size_t)Mimg * DM * 2);
  float*          kgFeatF  = (float*)alloc((size_t)Mkg * DM * 4);
  unsigned short* kgFeatB  = (unsigned short*)alloc((size_t)Mkg * DM * 2);

  unsigned short* Qbuf = (unsigned short*)alloc((size_t)Mkg * DM * 2);
  unsigned short* Kbuf = (unsigned short*)alloc((size_t)Mkg * DM * 2);
  unsigned short* Vbuf = (unsigned short*)alloc((size_t)Mkg * DM * 2);
  unsigned short* Obuf = (unsigned short*)alloc((size_t)Mkg * DM * 2);
  float*          attnF = (float*)alloc((size_t)Mkg * DM * 4);
  float*          xF    = (float*)alloc((size_t)Mkg * DM * 4);
  unsigned short* xB    = (unsigned short*)alloc((size_t)Mkg * DM * 2);
  unsigned short* hB    = (unsigned short*)alloc((size_t)Mkg * MIDD * 2);
  float*          h2F   = (float*)alloc((size_t)Mkg * DM * 4);
  float*          imgOutF = (float*)alloc((size_t)Mimg * DM * 4);
  float*          kgOutF  = (float*)alloc((size_t)Mkg * DM * 4);

  // ---- launchers ----
  auto cvt = [&](const float* s, unsigned short* d, int n) {
    int blocks = (n + 2047) / 2048;
    k_cvt_bf16<<<dim3(blocks), dim3(256), 0, stream>>>(s, d, n);
  };
  auto cvtT = [&](const float* s, unsigned short* d, int Kd, int Nd) {
    k_cvtT_bf16<<<dim3(Nd / 32, Kd / 32), dim3(256), 0, stream>>>(s, d, Kd, Nd);
  };
  auto gemm = [&](const unsigned short* A, const unsigned short* Bt, const float* bias,
                  float* Cf, unsigned short* Cb, int M, int N, int K, int relu) {
    k_gemm_bf16<<<dim3(N / 128, M / 128), dim3(256), 0, stream>>>(A, Bt, bias, Cf, Cb, M, N, K, relu);
  };

  // ---- weight transpose-convert / activation conversion ----
  cvt(question, q16, BB * DQ);
  cvtT(Wqi, wqi16, DQ, DM);
  cvtT(Wqk, wqk16, DQ, DM);
  auto cvtCross = [&](const CrossP& c, const CW& w) {
    cvtT(c.Wq, w.q, DM, DM);    cvtT(c.Wk, w.k, DM, DM);
    cvtT(c.Wv, w.v, DM, DM);    cvtT(c.Wo, w.o, DM, DM);
    cvtT(c.W1, w.w1, DM, MIDD); cvtT(c.W2, w.w2, MIDD, DM);
  };
  cvtCross(i2k, wi2k);
  cvtCross(k2i, wk2i);

  // ---- question projections (M = 128 -> one M-tile) ----
  gemm(q16, wqi16, bqi, qimgF, nullptr, BB, DM, DQ, 0);
  gemm(q16, wqk16, bqk, qkgF,  nullptr, BB, DM, DQ, 0);

  // ---- feature tanh (qg_att is identity -> nodes used directly) ----
  k_feat_tanh<<<dim3(Mimg / 8), dim3(256), 0, stream>>>(img_nodes, qimgF, imgFeatF, imgFeatB, NIMG);
  k_feat_tanh<<<dim3(Mkg  / 8), dim3(256), 0, stream>>>(kg_nodes,  qkgF,  kgFeatF,  kgFeatB,  NKG);

  // ---- cross blocks ----
  auto runCross = [&](const CrossP& cp, const CW& w,
                      const float* qfeatF, const unsigned short* qfeatB, int Nq,
                      const unsigned short* kvB, int Nk, float* outF) {
    const int Mq  = BB * Nq;
    const int Mkv = BB * Nk;
    gemm(qfeatB, w.q, cp.bq, nullptr, Qbuf, Mq,  DM, DM, 0);
    gemm(kvB,    w.k, cp.bk, nullptr, Kbuf, Mkv, DM, DM, 0);
    gemm(kvB,    w.v, cp.bv, nullptr, Vbuf, Mkv, DM, DM, 0);
    k_attn<<<dim3(Nq / 16, HN, BB), dim3(32), 0, stream>>>(Qbuf, Kbuf, Vbuf, Obuf, Nq, Nk);
    gemm(Obuf, w.o, cp.bo, attnF, nullptr, Mq, DM, DM, 0);
    k_ln512<<<dim3(Mq / 8), dim3(256), 0, stream>>>(qfeatF, attnF, cp.ln1g, cp.ln1b, xF, xB);
    gemm(xB, w.w1, cp.b1, nullptr, hB, Mq, MIDD, DM, 1);   // relu
    gemm(hB, w.w2, cp.b2, h2F, nullptr, Mq, DM, MIDD, 0);
    k_ln512<<<dim3(Mq / 8), dim3(256), 0, stream>>>(xF, h2F, cp.ln2g, cp.ln2b, outF, nullptr);
  };
  runCross(i2k, wi2k, imgFeatF, imgFeatB, NIMG, kgFeatB,  NKG,  imgOutF);
  runCross(k2i, wk2i, kgFeatF,  kgFeatB,  NKG,  imgFeatB, NIMG, kgOutF);

  // ---- residual + L2 normalize + concat into d_out ----
  float* out = (float*)d_out;
  k_res_l2<<<dim3(Mimg / 8), dim3(256), 0, stream>>>(img_nodes, imgOutF, out, NIMG, 0);
  k_res_l2<<<dim3(Mkg  / 8), dim3(256), 0, stream>>>(kg_nodes,  kgOutF,  out, NKG,  NIMG);
}
